// MultiHeadAttentionBlock_13039520710985
// MI455X (gfx1250) — compile-verified, matrix operand order checked by reading, not run on hardware
//
#include <hip/hip_runtime.h>
#include <hip/hip_bf16.h>

// ---------------------------------------------------------------------------
// MHA block for MI455X / gfx1250.
//  - bf16 WMMA (v_wmma_f32_16x16x32_bf16), fp32 accumulation
//  - one-shot f32->bf16 conversion pass (weights are L2-resident afterwards)
//  - 32x64 GEMM tiles per wave, 4 waves/block sharing the same N-strip
//  - flash attention with online softmax; K tiles double-buffered into LDS
//    via global_load_async_to_lds_b128 + s_wait_asynccnt (CDNA5 async path)
// ---------------------------------------------------------------------------

constexpr int Bc  = 2;
constexpr int Sc  = 2048;
constexpr int Dc  = 1024;
constexpr int Hc  = 16;
constexpr int DKc = 64;   // Dc / Hc

typedef __attribute__((ext_vector_type(16))) __bf16 v16bf;
typedef __attribute__((ext_vector_type(8)))  __bf16 v8bf;
typedef __attribute__((ext_vector_type(8)))  float  v8f;

// A/B operand K-mapping for 16-bit WMMA (ISA 7.12.2):
//   element e (0..15), khi = lane>>4:  K = (e<8 ? e : e+8) + khi*8
__device__ inline v16bf load_row_bf16(const __bf16* __restrict__ p, int khi) {
    v8bf lo = *(const v8bf*)(p + khi * 8);        // 16B aligned
    v8bf hi = *(const v8bf*)(p + 16 + khi * 8);
    v16bf a;
#pragma unroll
    for (int e = 0; e < 8; ++e) { a[e] = lo[e]; a[8 + e] = hi[e]; }
    return a;
}

__device__ inline v8f wmma_bf16(v16bf a, v16bf b, v8f c) {
    return __builtin_amdgcn_wmma_f32_16x16x32_bf16(false, a, false, b, (short)0, c,
                                                   false, false);
}

// ---------------------------------------------------------------------------
// f32 -> bf16 conversion pass (8 elems / thread, b128 in, b128 out)
// ---------------------------------------------------------------------------
__global__ __launch_bounds__(256) void cvt_f32_bf16(const float* __restrict__ x,
                                                    __bf16* __restrict__ y, int n8) {
    const int i = blockIdx.x * 256 + threadIdx.x;
    if (i >= n8) return;
    const float4 a = ((const float4*)x)[i * 2];
    const float4 b = ((const float4*)x)[i * 2 + 1];
    v8bf o;
    o[0] = (__bf16)a.x; o[1] = (__bf16)a.y; o[2] = (__bf16)a.z; o[3] = (__bf16)a.w;
    o[4] = (__bf16)b.x; o[5] = (__bf16)b.y; o[6] = (__bf16)b.z; o[7] = (__bf16)b.w;
    ((v8bf*)y)[i] = o;
}

// ---------------------------------------------------------------------------
// GEMM:  Y = X @ W^T + bias  (X: [M,K] bf16, W: [N,K] bf16)
//   OUT_MODE 0: bf16 [B,H,S,DK]    (Q / K for attention)
//   OUT_MODE 1: bf16 [B,H,DK,S]    (V transposed for attention)
//   OUT_MODE 2: fp32 row-major [M,N] (final output)
// One wave owns a 32(M) x 64(N) tile (8 WMMAs / K-step, B reused 2x).
// The 4 waves of a block share one N-strip -> B rows hit in L0/L2.
// ---------------------------------------------------------------------------
template <int OUT_MODE>
__global__ __launch_bounds__(128) void gemm32x64(const __bf16* __restrict__ X,
                                                 const __bf16* __restrict__ Wb,
                                                 const float* __restrict__ bias,
                                                 void* __restrict__ Yv) {
    constexpr int M  = Bc * Sc;   // 4096
    constexpr int N  = Dc;        // 1024
    constexpr int K  = Dc;        // 1024
    constexpr int NT = N / 64;    // 16 strips

    const int wave = threadIdx.x >> 5;
    const int lane = threadIdx.x & 31;
    const int tn4  = blockIdx.x % NT;               // shared by all 4 waves
    const int tm   = (blockIdx.x / NT) * 4 + wave;  // 0..127 (exact)
    const int khi  = lane >> 4;
    const int ml   = lane & 15;

    const __bf16* arow0 = X + (size_t)(tm * 32 + ml) * K;
    const __bf16* arow1 = arow0 + (size_t)16 * K;

    v8f acc[8] = {};

    for (int kk = 0; kk < K; kk += 32) {
        const v16bf a0 = load_row_bf16(arow0 + kk, khi);
        const v16bf a1 = load_row_bf16(arow1 + kk, khi);
        if (kk + 32 < K) {
            __builtin_prefetch(arow0 + kk + 32, 0, 0);
            __builtin_prefetch(arow1 + kk + 32, 0, 0);
        }
#pragma unroll
        for (int t = 0; t < 4; ++t) {
            const v16bf b =
                load_row_bf16(Wb + (size_t)(tn4 * 64 + t * 16 + ml) * K + kk, khi);
            acc[t]     = wmma_bf16(a0, b, acc[t]);
            acc[4 + t] = wmma_bf16(a1, b, acc[4 + t]);
        }
    }

#pragma unroll
    for (int t = 0; t < 4; ++t) {
        const int   ncol = tn4 * 64 + t * 16 + ml;
        const float bv   = bias[ncol];
#pragma unroll
        for (int half = 0; half < 2; ++half) {
#pragma unroll
            for (int r = 0; r < 8; ++r) {
                const int   m = tm * 32 + half * 16 + r + 8 * khi;
                const float y = acc[half * 4 + t][r] + bv;
                if constexpr (OUT_MODE == 2) {
                    ((float*)Yv)[(size_t)m * N + ncol] = y;
                } else {
                    const int bb = m / Sc, s = m % Sc;
                    const int h = ncol / DKc, dk = ncol % DKc;
                    size_t off;
                    if constexpr (OUT_MODE == 0)
                        off = (((size_t)bb * Hc + h) * Sc + s) * DKc + dk;
                    else
                        off = (((size_t)bb * Hc + h) * DKc + dk) * Sc + s;
                    ((__bf16*)Yv)[off] = (__bf16)y;
                }
            }
        }
    }
}

// ---------------------------------------------------------------------------
// Flash attention: one wave per (b*H+h, 16-query tile).
// K tiles (32 keys x 64 dk bf16 = 4KB) are double-buffered into LDS with
// global_load_async_to_lds_b128; consumer waits with s_wait_asynccnt.
// Exactly the last 32-key block needs causal masking -> split loop.
// ---------------------------------------------------------------------------
__global__ __launch_bounds__(32) void attn_fwd(const __bf16* __restrict__ Qw,
                                               const __bf16* __restrict__ Kw,
                                               const __bf16* __restrict__ Vt,
                                               __bf16* __restrict__ Ow) {
    __shared__ __bf16 kbuf[2][32 * DKc];   // 2 x 4KB double buffer
    __shared__ __bf16 pbuf[16 * 32];       // P tile C->A relayout

    const int lane = threadIdx.x & 31;
    const int khi  = lane >> 4;
    const int ml   = lane & 15;
    const int qt   = blockIdx.x % (Sc / 16);
    const int bh   = blockIdx.x / (Sc / 16);

    const __bf16* Qb = Qw + (size_t)bh * Sc * DKc;
    const __bf16* Kb = Kw + (size_t)bh * Sc * DKc;
    const __bf16* Vb = Vt + (size_t)bh * DKc * Sc;

    const int q_base = qt * 16;

    // Q tile as WMMA-A, DK=64 -> two K=32 slices, held in VGPRs for the loop
    const v16bf a0 = load_row_bf16(Qb + (size_t)(q_base + ml) * DKc + 0, khi);
    const v16bf a1 = load_row_bf16(Qb + (size_t)(q_base + ml) * DKc + 32, khi);

    float mrun[8], lrun[8];
#pragma unroll
    for (int r = 0; r < 8; ++r) { mrun[r] = -__builtin_inff(); lrun[r] = 0.f; }
    v8f o[4] = {};

    const float scale = 0.125f;                    // 1/sqrt(64)
    const int   kend  = q_base + 16;               // causal: keys [0, kend)
    const int   nblk  = (kend + 31) / 32;          // last block is the masked one

    // ---- async stage of one 32x64 bf16 K tile into LDS -----------------
    auto stage = [&](int bufi, int k0) {
#pragma unroll
        for (int j = 0; j < 8; ++j) {
            const int     e = (lane + 32 * j) * 8;          // 8 bf16 = 16B chunk
            const __bf16* g = Kb + (size_t)k0 * DKc + e;    // rows contiguous
            const unsigned l = (unsigned)(uintptr_t)&kbuf[bufi][e]; // LDS addr
            asm volatile("global_load_async_to_lds_b128 %0, %1, off"
                         :: "v"(l), "v"(g)
                         : "memory");
        }
    };

    stage(0, 0);

    for (int blk = 0; blk < nblk; ++blk) {
        const int k0 = blk * 32;
        if (blk + 1 < nblk) {
            stage((blk + 1) & 1, k0 + 32);                  // prefetch next tile
            asm volatile("s_wait_asynccnt 0x1" ::: "memory"); // current tile ready
        } else {
            asm volatile("s_wait_asynccnt 0x0" ::: "memory");
        }
        const __bf16* kt = kbuf[blk & 1];

        // ---- scores for two 16-key subtiles (from LDS) -----------------
        v8f s0 = {}, s1 = {};
        {
            v16bf b0 = load_row_bf16(kt + (size_t)ml * DKc + 0, khi);
            v16bf b1 = load_row_bf16(kt + (size_t)ml * DKc + 32, khi);
            s0 = wmma_bf16(a0, b0, s0);
            s0 = wmma_bf16(a1, b1, s0);
            b0 = load_row_bf16(kt + (size_t)(16 + ml) * DKc + 0, khi);
            b1 = load_row_bf16(kt + (size_t)(16 + ml) * DKc + 32, khi);
            s1 = wmma_bf16(a0, b0, s1);
            s1 = wmma_bf16(a1, b1, s1);
        }

        // ---- (masked only on the last block) online softmax ------------
        const bool masked = (blk == nblk - 1);
        const int  ncol0  = k0 + ml;
        const int  ncol1  = ncol0 + 16;
        float pr0[8], pr1[8], fac[8];
#pragma unroll
        for (int r = 0; r < 8; ++r) {
            const int qi = q_base + r + 8 * khi;
            float v0 = s0[r] * scale;
            float v1 = s1[r] * scale;
            if (masked) {                      // wave-uniform branch
                if (ncol0 > qi) v0 = -__builtin_inff();
                if (ncol1 > qi) v1 = -__builtin_inff();
            }
            float mx = fmaxf(v0, v1);
#pragma unroll
            for (int off = 1; off < 16; off <<= 1)   // reduce across 16-lane half
                mx = fmaxf(mx, __shfl_xor(mx, off, 32));
            const float mnew = fmaxf(mrun[r], mx);
            const float p0 = __expf(v0 - mnew);
            const float p1 = __expf(v1 - mnew);
            float sum = p0 + p1;
#pragma unroll
            for (int off = 1; off < 16; off <<= 1)
                sum += __shfl_xor(sum, off, 32);
            fac[r]  = __expf(mrun[r] - mnew);
            lrun[r] = lrun[r] * fac[r] + sum;
            mrun[r] = mnew;
            pr0[r] = p0; pr1[r] = p1;
        }

        // rescale running O
#pragma unroll
        for (int t = 0; t < 4; ++t)
#pragma unroll
            for (int r = 0; r < 8; ++r) o[t][r] *= fac[r];

        // ---- P: C-layout -> A-layout via LDS ---------------------------
        __syncthreads();
#pragma unroll
        for (int r = 0; r < 8; ++r) {
            const int m = r + 8 * khi;
            pbuf[m * 32 + ml]      = (__bf16)pr0[r];
            pbuf[m * 32 + 16 + ml] = (__bf16)pr1[r];
        }
        __syncthreads();
        v16bf pa;
        {
            v8bf lo = *(const v8bf*)&pbuf[ml * 32 + khi * 8];
            v8bf hi = *(const v8bf*)&pbuf[ml * 32 + 16 + khi * 8];
#pragma unroll
            for (int e = 0; e < 8; ++e) { pa[e] = lo[e]; pa[8 + e] = hi[e]; }
        }

        // ---- O += P @ V  (V transposed: contiguous along key dim) ------
#pragma unroll
        for (int t = 0; t < 4; ++t) {
            const __bf16* vrow = Vb + (size_t)(t * 16 + ml) * Sc + k0;
            const v16bf vb = load_row_bf16(vrow, khi);
            o[t] = wmma_bf16(pa, vb, o[t]);
        }
    }

    // ---- normalize + write O as bf16 [B,S,D] ---------------------------
    const int bb = bh / Hc, h = bh % Hc;
    float rl[8];
#pragma unroll
    for (int r = 0; r < 8; ++r) rl[r] = 1.0f / lrun[r];
#pragma unroll
    for (int t = 0; t < 4; ++t)
#pragma unroll
        for (int r = 0; r < 8; ++r) {
            const int   qi  = q_base + r + 8 * khi;
            const float val = o[t][r] * rl[r];
            const size_t off =
                ((size_t)bb * Sc + qi) * Dc + h * DKc + t * 16 + ml;
            Ow[off] = (__bf16)val;
        }
}

// ---------------------------------------------------------------------------
extern "C" void kernel_launch(void* const* d_in, const int* in_sizes, int n_in,
                              void* d_out, int out_size, void* d_ws, size_t ws_size,
                              hipStream_t stream) {
    const float* q  = (const float*)d_in[0];
    const float* k  = (const float*)d_in[1];
    const float* v  = (const float*)d_in[2];
    // d_in[3] = mask (causal handled analytically)
    const float* Wq = (const float*)d_in[4];
    const float* bq = (const float*)d_in[5];
    const float* Wk = (const float*)d_in[6];
    const float* bk = (const float*)d_in[7];
    const float* Wv = (const float*)d_in[8];
    const float* bv = (const float*)d_in[9];
    const float* Wo = (const float*)d_in[10];
    const float* bo = (const float*)d_in[11];
    float* out = (float*)d_out;

    const size_t sz  = (size_t)Bc * Sc * Dc;   // 4Mi elements (activations)
    const size_t wsz = (size_t)Dc * Dc;        // 1Mi elements (weights)

    __bf16* p  = (__bf16*)d_ws;
    __bf16* qb  = p; p += sz;   // [B,H,S,DK]
    __bf16* kb  = p; p += sz;   // [B,H,S,DK]
    __bf16* vt  = p; p += sz;   // [B,H,DK,S]
    __bf16* ob  = p; p += sz;   // [B,S,D]
    __bf16* xq  = p; p += sz;   // bf16 copies of inputs
    __bf16* xk  = p; p += sz;
    __bf16* xv  = p; p += sz;
    __bf16* wqb = p; p += wsz;  // bf16 copies of weights
    __bf16* wkb = p; p += wsz;
    __bf16* wvb = p; p += wsz;
    __bf16* wob = p; p += wsz;

    // ---- conversion pass ----------------------------------------------
    {
        const int na8 = (int)(sz / 8), nw8 = (int)(wsz / 8);
        dim3 ga((na8 + 255) / 256), gw((nw8 + 255) / 256), blk(256);
        cvt_f32_bf16<<<ga, blk, 0, stream>>>(q, xq, na8);
        cvt_f32_bf16<<<ga, blk, 0, stream>>>(k, xk, na8);
        cvt_f32_bf16<<<ga, blk, 0, stream>>>(v, xv, na8);
        cvt_f32_bf16<<<gw, blk, 0, stream>>>(Wq, wqb, nw8);
        cvt_f32_bf16<<<gw, blk, 0, stream>>>(Wk, wkb, nw8);
        cvt_f32_bf16<<<gw, blk, 0, stream>>>(Wv, wvb, nw8);
        cvt_f32_bf16<<<gw, blk, 0, stream>>>(Wo, wob, nw8);
    }

    // ---- projections ---------------------------------------------------
    constexpr int NT = Dc / 64;                      // 16
    dim3 ggrid((Bc * Sc / 32) * NT / 4), gblk(128);  // 512 blocks, 4 waves each
    gemm32x64<0><<<ggrid, gblk, 0, stream>>>(xq, wqb, bq, qb);
    gemm32x64<0><<<ggrid, gblk, 0, stream>>>(xk, wkb, bk, kb);
    gemm32x64<1><<<ggrid, gblk, 0, stream>>>(xv, wvb, bv, vt);

    // ---- attention -----------------------------------------------------
    attn_fwd<<<Bc * Hc * (Sc / 16), 32, 0, stream>>>(qb, kb, vt, ob);

    // ---- output projection --------------------------------------------
    gemm32x64<2><<<ggrid, gblk, 0, stream>>>(ob, wob, bo, out);
}
